// LGESQL_2224793059899
// MI455X (gfx1250) — matrix operand order repaired and checked by reading, not compiled, not run on hardware
//
#include <hip/hip_runtime.h>
#include <hip/hip_bf16.h>

// ---------------------------------------------------------------------------
// LGESQL relational graph transformer for MI455X (gfx1250, wave32, WMMA).
// GEMMs on v_wmma_f32_16x16x32_f16 (f16 in, f32 accumulate).
// Weights pre-converted to f16 column-major scratch so B fragments are
// contiguous 32-byte loads (global_load_b128) with no cvt in the hot loop.
// FFN (128->512->128) fused through LDS; last-layer edge update is dead code
// in the reference and skipped.
// ---------------------------------------------------------------------------

typedef __attribute__((ext_vector_type(16))) _Float16 v16h;
typedef __attribute__((ext_vector_type(8)))  float    v8f;

#define N_NODES 50000
#define D_DIM   128
#define H_HEADS 8
#define DK_DIM  16
#define L_LAYERS 2
#define EG_EDGES 800000
#define EL_EDGES 400000
#define ELG_EDGES 1200000
#define DFF_DIM 512

// LDS strides (in elements) chosen to keep 16-byte alignment of 8-f16 chunks.
#define ASTR 136   // f16 stride for a 16x128 A tile (128+8)
#define HSTR 528   // f16 stride for the 16x512 hidden tile (512+16)
#define OSTR 132   // f32 stride for the 16x128 output tile

// --------------------------- device helpers --------------------------------

// A-fragment (16x32 f16, contiguous k = 16*half + e) from an LDS f16 row.
__device__ __forceinline__ v16h frag_a_lds(const _Float16* __restrict__ row,
                                           int k0, int half) {
  const _Float16* p = row + k0 + half * 16;
  v16h a;
#pragma unroll
  for (int e = 0; e < 8; ++e) { a[e] = p[e]; a[e + 8] = p[e + 8]; }
  return a;
}

// B-fragment (32x16) from f16 column-major Wt[n][k] (ldk = K).
// 16 contiguous f16 = 32 bytes -> 2x global_load_b128, no conversions.
__device__ __forceinline__ v16h frag_b_t(const _Float16* __restrict__ Wt,
                                         int K, int ncol, int k0, int half) {
  return *reinterpret_cast<const v16h*>(Wt + (long)ncol * K + k0 + half * 16);
}

__device__ __forceinline__ float wave_reduce_add(float v) {
#pragma unroll
  for (int off = 16; off > 0; off >>= 1) v += __shfl_xor(v, off, 32);
  return v;
}

// --------------------------- utility kernels -------------------------------

__global__ void k_zero(float* __restrict__ p, long n) {
  long i = (long)blockIdx.x * blockDim.x + threadIdx.x;
  long stride = (long)gridDim.x * blockDim.x;
  for (; i < n; i += stride) p[i] = 0.0f;
}

// Wt[n][k] = (f16) W[k][n]  -- per-GEMM weight transpose+convert (tiny)
__global__ void k_cvt_wT(const float* __restrict__ W,
                         _Float16* __restrict__ Wt, int K, int NC) {
  int t = blockIdx.x * blockDim.x + threadIdx.x;
  if (t >= K * NC) return;
  int k = t / NC, n = t % NC;
  Wt[(long)n * K + k] = (_Float16)W[t];
}

// lgx[e] = rel_emb[global_edges[local_idx[e]]]   (float4 granularity)
__global__ void k_gather_lgx(const float4* __restrict__ rel,
                             const int* __restrict__ ge,
                             const int* __restrict__ li,
                             float4* __restrict__ lgx, int nE) {
  int t = blockIdx.x * blockDim.x + threadIdx.x;
  if (t >= nE * 32) return;
  int e = t >> 5, c = t & 31;
  int r = ge[li[e]];
  lgx[t] = rel[(long)r * 32 + c];
}

// out[e] += x[idx[e]]  (float4 granularity, rows of 128 floats)
__global__ void k_add_gather(float4* __restrict__ out,
                             const float4* __restrict__ x,
                             const int* __restrict__ idx, int nE) {
  int t = blockIdx.x * blockDim.x + threadIdx.x;
  if (t >= nE * 32) return;
  int e = t >> 5, c = t & 31;
  float4 a = out[t];
  float4 b = x[(long)idx[e] * 32 + c];
  a.x += b.x; a.y += b.y; a.z += b.z; a.w += b.w;
  out[t] = a;
}

// Edge softmax numerator/denominator scatter (segment_sum via f32 atomics).
// k_e = k[src]+eadd, v_e = v[src]+eadd (eadd nullable), query = q[dst].
__global__ void k_attn(const float* __restrict__ q, const float* __restrict__ k,
                       const float* __restrict__ v,
                       const float* __restrict__ eadd,
                       const int* __restrict__ src, const int* __restrict__ dst,
                       float* __restrict__ wv, float* __restrict__ z, int nE) {
  int t = blockIdx.x * blockDim.x + threadIdx.x;
  if (t >= nE * H_HEADS) return;
  int e = t >> 3, h = t & 7;
  int s = src[e], d = dst[e];
  const float* kp = k + (long)s * D_DIM + h * DK_DIM;
  const float* vp = v + (long)s * D_DIM + h * DK_DIM;
  const float* qp = q + (long)d * D_DIM + h * DK_DIM;
  const float* ep = eadd ? eadd + (long)e * D_DIM + h * DK_DIM : nullptr;
  float dot = 0.0f;
  if (ep) {
#pragma unroll
    for (int i = 0; i < DK_DIM; ++i) dot += (kp[i] + ep[i]) * qp[i];
  } else {
#pragma unroll
    for (int i = 0; i < DK_DIM; ++i) dot += kp[i] * qp[i];
  }
  float sc = fminf(fmaxf(dot * 0.25f, -10.0f), 10.0f);  // /sqrt(DK)=4, clamp
  float sv = expf(sc);
  float* wp = wv + (long)d * D_DIM + h * DK_DIM;
  if (ep) {
#pragma unroll
    for (int i = 0; i < DK_DIM; ++i) atomicAdd(&wp[i], sv * (vp[i] + ep[i]));
  } else {
#pragma unroll
    for (int i = 0; i < DK_DIM; ++i) atomicAdd(&wp[i], sv * vp[i]);
  }
  atomicAdd(&z[(long)d * H_HEADS + h], sv);
}

// wv /= where(z>0, z, 1), in place
__global__ void k_divz(float* __restrict__ wv, const float* __restrict__ z,
                       long n) {
  long t = (long)blockIdx.x * blockDim.x + threadIdx.x;
  if (t >= n) return;
  long row = t >> 7;
  int h = (int)((t >> 4) & 7);
  float zz = z[row * H_HEADS + h];
  wv[t] /= (zz > 0.0f ? zz : 1.0f);
}

// Row LayerNorm over 128 cols, in place. One wave per row.
__global__ void k_ln(float* __restrict__ T, const float* __restrict__ g,
                     const float* __restrict__ b, int M) {
  int lane = threadIdx.x & 31, wave = threadIdx.x >> 5;
  long row = (long)blockIdx.x * 8 + wave;
  if (row >= M) return;
  float vals[4];
#pragma unroll
  for (int m = 0; m < 4; ++m) vals[m] = T[row * D_DIM + lane + 32 * m];
  float s = wave_reduce_add(vals[0] + vals[1] + vals[2] + vals[3]);
  float mean = s * (1.0f / 128.0f);
  float vs = 0.0f;
#pragma unroll
  for (int m = 0; m < 4; ++m) { float d = vals[m] - mean; vs += d * d; }
  vs = wave_reduce_add(vs);
  float rsq = rsqrtf(vs * (1.0f / 128.0f) + 1e-5f);
#pragma unroll
  for (int m = 0; m < 4; ++m) {
    int c = lane + 32 * m;
    T[row * D_DIM + c] = (vals[m] - mean) * rsq * g[c] + b[c];
  }
}

// --------------------------- WMMA GEMM -------------------------------------
// C[M x 128] = A[M x K] @ Wt^T (+bias) (+residual), Wt is f16 [128][K].
// One block = 8 waves = one 16-row stripe across all 8 column tiles.
// A tile staged once in LDS as f16 and shared by all 8 waves.
__global__ __launch_bounds__(256) void k_gemm(
    const float* __restrict__ A, const _Float16* __restrict__ Wt,
    const float* __restrict__ bias, const float* __restrict__ res,
    float* __restrict__ C, int M, int K) {
  __shared__ _Float16 abuf[16 * ASTR];  // supports K <= 128
  const int lane = threadIdx.x & 31;
  const int wave = threadIdx.x >> 5;
  const int half = lane >> 4;
  const int lid = lane & 15;
  const long rowbase = (long)blockIdx.x * 16;
  const int ncol = wave * 16 + lid;

  for (int i = threadIdx.x; i < 16 * K; i += 256) {
    int r = i / K, c = i % K;
    abuf[r * ASTR + c] = (_Float16)A[(rowbase + r) * K + c];
  }
  __syncthreads();

  v8f acc = {};
  for (int k0 = 0; k0 < K; k0 += 32) {
    v16h a = frag_a_lds(abuf + lid * ASTR, k0, half);
    v16h b = frag_b_t(Wt, K, ncol, k0, half);
    acc = __builtin_amdgcn_wmma_f32_16x16x32_f16(false, a, false, b, (short)0,
                                                 acc, false, false);
  }
  const float bv = bias ? bias[ncol] : 0.0f;
#pragma unroll
  for (int j = 0; j < 8; ++j) {
    long row = rowbase + j + (half << 3);
    float v = acc[j] + bv;
    if (res) v += res[row * D_DIM + ncol];
    C[row * D_DIM + ncol] = v;
  }
}

// --------------------------- fused FFN + LN --------------------------------
// T = LN(T + relu(T@W1+b1)@W2 + b2), 128->512->128, in place.
// Wt1 f16 [512][128], Wt2 f16 [128][512] (column-major of W1/W2).
// One block handles 16 rows; hidden 16x512 tile lives only in LDS (f16).
__global__ __launch_bounds__(256) void k_ffn(
    float* __restrict__ T, const _Float16* __restrict__ Wt1,
    const float* __restrict__ b1, const _Float16* __restrict__ Wt2,
    const float* __restrict__ b2, const float* __restrict__ g,
    const float* __restrict__ bb) {
  __shared__ _Float16 abuf[16 * ASTR];
  __shared__ _Float16 hbuf[16 * HSTR];
  __shared__ float    obuf[16 * OSTR];
  const int lane = threadIdx.x & 31;
  const int wave = threadIdx.x >> 5;
  const int half = lane >> 4;
  const int lid = lane & 15;
  const long rowbase = (long)blockIdx.x * 16;

  // stage the 16x128 input tile once
  for (int i = threadIdx.x; i < 16 * D_DIM; i += 256) {
    int r = i >> 7, c = i & 127;
    abuf[r * ASTR + c] = (_Float16)T[(rowbase + r) * D_DIM + c];
  }
  __syncthreads();

  // phase 1: h = relu(T@W1 + b1), 32 column tiles, 4 per wave
#pragma unroll
  for (int t = 0; t < 4; ++t) {
    const int ncol = (wave * 4 + t) * 16 + lid;
    v8f acc = {};
    for (int k0 = 0; k0 < D_DIM; k0 += 32) {
      v16h a = frag_a_lds(abuf + lid * ASTR, k0, half);
      v16h b = frag_b_t(Wt1, D_DIM, ncol, k0, half);
      acc = __builtin_amdgcn_wmma_f32_16x16x32_f16(false, a, false, b,
                                                   (short)0, acc, false, false);
    }
    const float bv = b1[ncol];
#pragma unroll
    for (int j = 0; j < 8; ++j) {
      int rl = j + (half << 3);
      float v = acc[j] + bv;
      hbuf[rl * HSTR + ncol] = (_Float16)(v > 0.0f ? v : 0.0f);
    }
  }
  __syncthreads();

  // phase 2: out = h@W2 + b2 + T (residual), one column tile per wave
  {
    const int ncol = wave * 16 + lid;
    v8f acc = {};
    for (int k0 = 0; k0 < DFF_DIM; k0 += 32) {
      v16h a = frag_a_lds(hbuf + lid * HSTR, k0, half);
      v16h b = frag_b_t(Wt2, DFF_DIM, ncol, k0, half);
      acc = __builtin_amdgcn_wmma_f32_16x16x32_f16(false, a, false, b,
                                                   (short)0, acc, false, false);
    }
    const float bv = b2[ncol];
#pragma unroll
    for (int j = 0; j < 8; ++j) {
      int rl = j + (half << 3);
      obuf[rl * OSTR + ncol] = acc[j] + bv + T[(rowbase + rl) * D_DIM + ncol];
    }
  }
  __syncthreads();

  // LayerNorm: 8 waves x 2 rows
#pragma unroll
  for (int rr = 0; rr < 2; ++rr) {
    const int rl = wave * 2 + rr;
    float vals[4];
#pragma unroll
    for (int m = 0; m < 4; ++m) vals[m] = obuf[rl * OSTR + lane + 32 * m];
    float s = wave_reduce_add(vals[0] + vals[1] + vals[2] + vals[3]);
    float mean = s * (1.0f / 128.0f);
    float vs = 0.0f;
#pragma unroll
    for (int m = 0; m < 4; ++m) { float d = vals[m] - mean; vs += d * d; }
    vs = wave_reduce_add(vs);
    float rsq = rsqrtf(vs * (1.0f / 128.0f) + 1e-5f);
#pragma unroll
    for (int m = 0; m < 4; ++m) {
      int c = lane + 32 * m;
      T[(rowbase + rl) * D_DIM + c] = (vals[m] - mean) * rsq * g[c] + bb[c];
    }
  }
}

// ------------------------------- host --------------------------------------

extern "C" void kernel_launch(void* const* d_in, const int* in_sizes, int n_in,
                              void* d_out, int out_size, void* d_ws,
                              size_t ws_size, hipStream_t stream) {
  const float* x_in   = (const float*)d_in[0];
  const float* rel    = (const float*)d_in[1];
  const float* nWq = (const float*)d_in[2],  *nbq = (const float*)d_in[3];
  const float* nWk = (const float*)d_in[4],  *nWv = (const float*)d_in[5];
  const float* nWo = (const float*)d_in[6],  *nbo = (const float*)d_in[7];
  const float* nlg = (const float*)d_in[8],  *nlb = (const float*)d_in[9];
  const float* nW1 = (const float*)d_in[10], *nb1 = (const float*)d_in[11];
  const float* nW2 = (const float*)d_in[12], *nb2 = (const float*)d_in[13];
  const float* nl2g = (const float*)d_in[14], *nl2b = (const float*)d_in[15];
  const float* eWq = (const float*)d_in[16], *ebq = (const float*)d_in[17];
  const float* eWk = (const float*)d_in[18], *eWv = (const float*)d_in[19];
  const float* eWo = (const float*)d_in[20], *ebo = (const float*)d_in[21];
  const float* elg = (const float*)d_in[22], *elb = (const float*)d_in[23];
  const float* eW1 = (const float*)d_in[24], *eb1 = (const float*)d_in[25];
  const float* eW2 = (const float*)d_in[26], *eb2 = (const float*)d_in[27];
  const float* el2g = (const float*)d_in[28], *el2b = (const float*)d_in[29];
  const int* g_edges = (const int*)d_in[30];
  const int* l_idx   = (const int*)d_in[31];
  const int* g_src   = (const int*)d_in[32];
  const int* g_dst   = (const int*)d_in[33];
  const int* lg_src  = (const int*)d_in[34];
  const int* lg_dst  = (const int*)d_in[35];
  float* out = (float*)d_out;

  // workspace carve-out
  float* ws = (float*)d_ws;
  size_t o = 0;
  float* lgx = ws + o;  o += (size_t)EL_EDGES * D_DIM;
  float* bA  = ws + o;  o += (size_t)EL_EDGES * D_DIM;   // q / qe
  float* bB  = ws + o;  o += (size_t)EL_EDGES * D_DIM;   // k / ke
  float* bC  = ws + o;  o += (size_t)EL_EDGES * D_DIM;   // v / ve
  float* bWV = ws + o;  o += (size_t)EL_EDGES * D_DIM;   // softmax numer / o
  float* bZ  = ws + o;  o += (size_t)EL_EDGES * H_HEADS; // softmax denom
  float* bX  = ws + o;  o += (size_t)N_NODES * D_DIM;    // layer-0 out_x
  _Float16* wT1 = (_Float16*)(ws + o);                   // f16 weight scratch
  o += (size_t)(D_DIM * DFF_DIM) / 2;                    //   (as float slots)
  _Float16* wT2 = (_Float16*)(ws + o);
  o += (size_t)(DFF_DIM * D_DIM) / 2;

  const int T256 = 256;
  const int CVT_DD = (D_DIM * D_DIM + 255) / 256;          // 64 blocks
  const int CVT_DF = (D_DIM * DFF_DIM + 255) / 256;        // 256 blocks

  // lgx = rel_emb[global_edges[local_idx]]
  k_gather_lgx<<<(EL_EDGES * 32 + 255) / 256, T256, 0, stream>>>(
      (const float4*)rel, g_edges, l_idx, (float4*)lgx, EL_EDGES);

  for (int i = 0; i < L_LAYERS; ++i) {
    const float* xin = (i == 0) ? x_in : bX;
    float* xout = (i == 0) ? bX : out;
    const size_t wD = (size_t)i * D_DIM * D_DIM;
    const size_t vD = (size_t)i * D_DIM;
    const size_t w1 = (size_t)i * D_DIM * DFF_DIM;
    const size_t vF = (size_t)i * DFF_DIM;

    // ---------- node update ----------
    k_cvt_wT<<<CVT_DD, T256, 0, stream>>>(nWq + wD, wT1, D_DIM, D_DIM);
    k_gemm<<<N_NODES / 16, T256, 0, stream>>>(xin, wT1, nbq + vD, nullptr, bA,
                                              N_NODES, D_DIM);
    k_cvt_wT<<<CVT_DD, T256, 0, stream>>>(nWk + wD, wT1, D_DIM, D_DIM);
    k_gemm<<<N_NODES / 16, T256, 0, stream>>>(xin, wT1, nullptr, nullptr, bB,
                                              N_NODES, D_DIM);
    k_cvt_wT<<<CVT_DD, T256, 0, stream>>>(nWv + wD, wT1, D_DIM, D_DIM);
    k_gemm<<<N_NODES / 16, T256, 0, stream>>>(xin, wT1, nullptr, nullptr, bC,
                                              N_NODES, D_DIM);
    k_zero<<<4096, T256, 0, stream>>>(bWV, (long)N_NODES * D_DIM);
    k_zero<<<4096, T256, 0, stream>>>(bZ, (long)N_NODES * H_HEADS);
    k_attn<<<(EL_EDGES * H_HEADS + 255) / 256, T256, 0, stream>>>(
        bA, bB, bC, lgx, g_src, g_dst, bWV, bZ, EL_EDGES);
    k_divz<<<(N_NODES * D_DIM + 255) / 256, T256, 0, stream>>>(
        bWV, bZ, (long)N_NODES * D_DIM);
    k_cvt_wT<<<CVT_DD, T256, 0, stream>>>(nWo + wD, wT1, D_DIM, D_DIM);
    k_gemm<<<N_NODES / 16, T256, 0, stream>>>(bWV, wT1, nbo + vD, xin, xout,
                                              N_NODES, D_DIM);
    k_ln<<<N_NODES / 8, T256, 0, stream>>>(xout, nlg + vD, nlb + vD, N_NODES);
    k_cvt_wT<<<CVT_DF, T256, 0, stream>>>(nW1 + w1, wT1, D_DIM, DFF_DIM);
    k_cvt_wT<<<CVT_DF, T256, 0, stream>>>(nW2 + w1, wT2, DFF_DIM, D_DIM);
    k_ffn<<<N_NODES / 16, T256, 0, stream>>>(xout, wT1, nb1 + vF, wT2,
                                             nb2 + vD, nl2g + vD, nl2b + vD);

    // ---------- edge update (dead in the last layer: lgx unused after) ----
    if (i < L_LAYERS - 1) {
      k_cvt_wT<<<CVT_DD, T256, 0, stream>>>(eWq + wD, wT1, D_DIM, D_DIM);
      k_gemm<<<EL_EDGES / 16, T256, 0, stream>>>(lgx, wT1, ebq + vD, nullptr,
                                                 bA, EL_EDGES, D_DIM);
      k_add_gather<<<(EL_EDGES * 32 + 255) / 256, T256, 0, stream>>>(
          (float4*)bA, (const float4*)xin, g_src, EL_EDGES);  // + src_x
      k_cvt_wT<<<CVT_DD, T256, 0, stream>>>(eWk + wD, wT1, D_DIM, D_DIM);
      k_gemm<<<EL_EDGES / 16, T256, 0, stream>>>(lgx, wT1, nullptr, nullptr,
                                                 bB, EL_EDGES, D_DIM);
      k_cvt_wT<<<CVT_DD, T256, 0, stream>>>(eWv + wD, wT1, D_DIM, D_DIM);
      k_gemm<<<EL_EDGES / 16, T256, 0, stream>>>(lgx, wT1, nullptr, nullptr,
                                                 bC, EL_EDGES, D_DIM);
      k_add_gather<<<(EL_EDGES * 32 + 255) / 256, T256, 0, stream>>>(
          (float4*)bC, (const float4*)xin, g_dst, EL_EDGES);  // + dst_x
      k_zero<<<4096, T256, 0, stream>>>(bWV, (long)EL_EDGES * D_DIM);
      k_zero<<<4096, T256, 0, stream>>>(bZ, (long)EL_EDGES * H_HEADS);
      k_attn<<<(ELG_EDGES * H_HEADS + 255) / 256, T256, 0, stream>>>(
          bA, bB, bC, nullptr, lg_src, lg_dst, bWV, bZ, ELG_EDGES);
      k_divz<<<((long)EL_EDGES * D_DIM + 255) / 256, T256, 0, stream>>>(
          bWV, bZ, (long)EL_EDGES * D_DIM);
      k_cvt_wT<<<CVT_DD, T256, 0, stream>>>(eWo + wD, wT1, D_DIM, D_DIM);
      k_gemm<<<EL_EDGES / 16, T256, 0, stream>>>(bWV, wT1, ebo + vD, lgx, lgx,
                                                 EL_EDGES, D_DIM);
      k_ln<<<EL_EDGES / 8, T256, 0, stream>>>(lgx, elg + vD, elb + vD,
                                              EL_EDGES);
      k_cvt_wT<<<CVT_DF, T256, 0, stream>>>(eW1 + w1, wT1, D_DIM, DFF_DIM);
      k_cvt_wT<<<CVT_DF, T256, 0, stream>>>(eW2 + w1, wT2, DFF_DIM, D_DIM);
      k_ffn<<<EL_EDGES / 16, T256, 0, stream>>>(lgx, wT1, eb1 + vF, wT2,
                                                eb2 + vD, el2g + vD,
                                                el2b + vD);
    }
  }
}
